// MiniGPT_6657199309175
// MI455X (gfx1250) — compile-verified, matrix operand order checked by reading, not run on hardware
//
#include <hip/hip_runtime.h>

// ---------------------------------------------------------------------------
// MiniGPT forward for MI455X (gfx1250, wave32, WMMA bf16 16x16x32, f32 acc)
// ---------------------------------------------------------------------------
#define E_DIM 1024
#define T_DIM 1024
#define B_DIM 4
#define H_DIM 16
#define D_DIM 64
#define L_DIM 8
#define M_ROWS (B_DIM * T_DIM) // 4096 token rows

typedef __attribute__((ext_vector_type(16))) __bf16 bf16x16;
typedef __attribute__((ext_vector_type(8)))  float  floatx8;

// --- f32 -> bf16 (RNE) pack: single v_cvt_pk_bf16_f32 per 2 elements. ------
// The clang builtin is not declared by this toolchain (probe round 2), but
// the instruction exists in the CDNA5 VALU; emit it via inline asm.
// dst.lo16 = bf16(a), dst.hi16 = bf16(b).
__device__ __forceinline__ unsigned int pk2(float a, float b) {
  unsigned int r;
  asm("v_cvt_pk_bf16_f32 %0, %1, %2" : "=v"(r) : "v"(a), "v"(b));
  return r;
}
// Single value: low half of the packed pair (ds_store_b16 takes low 16 bits).
__device__ __forceinline__ unsigned short f2bf(float f) {
  return (unsigned short)pk2(f, f);
}

// --- 16-lane butterfly reductions via DPP16 (no LDS traffic). ---------------
template <int CTRL>
__device__ __forceinline__ float dpp_mov(float x) {
  int i = __builtin_amdgcn_mov_dpp(__builtin_bit_cast(int, x), CTRL, 0xF, 0xF, true);
  return __builtin_bit_cast(float, i);
}
__device__ __forceinline__ float redmax16(float x) {
  x = fmaxf(x, dpp_mov<0xB1>(x));   // quad_perm xor1
  x = fmaxf(x, dpp_mov<0x4E>(x));   // quad_perm xor2
  x = fmaxf(x, dpp_mov<0x141>(x));  // row_half_mirror (xor4 over equal quads)
  x = fmaxf(x, dpp_mov<0x140>(x));  // row_mirror      (xor8 over equal octs)
  return x;
}
__device__ __forceinline__ float redsum16(float x) {
  x += dpp_mov<0xB1>(x);
  x += dpp_mov<0x4E>(x);
  x += dpp_mov<0x141>(x);
  x += dpp_mov<0x140>(x);
  return x;
}

// Load a 16x32 bf16 A/B fragment from LDS following ISA 7.12.2:
// lanes 0-15 hold K {0..7, 16..23}, lanes 16-31 hold K {8..15, 24..31};
// row index = lane&15 (+ tile offset). Two 16B ds_load_b128 per fragment.
__device__ __forceinline__ bf16x16 frag_ld(const unsigned short* base, int stride,
                                           int row, int kbase, int hf) {
  const unsigned short* p = base + row * stride + kbase + hf * 8;
  union { uint4 u[2]; bf16x16 v; } r;
  r.u[0] = *(const uint4*)(p);
  r.u[1] = *(const uint4*)(p + 16);
  return r.v;
}

__device__ __forceinline__ floatx8 wmma_bf16(bf16x16 a, bf16x16 b, floatx8 c) {
  return __builtin_amdgcn_wmma_f32_16x16x32_bf16(false, a, false, b,
                                                 (short)0, c, false, false);
}

// ---------------------------------------------------------------------------
// Embedding: x[b*T+t, :] = tok_emb[idx[b,t], :] + pos_emb[t, :]
// ---------------------------------------------------------------------------
__global__ __launch_bounds__(256)
void embed_kernel(const int* __restrict__ idx, const float* __restrict__ tok,
                  const float* __restrict__ pos, float* __restrict__ x) {
  const int row = blockIdx.x;           // 0..4095
  const int t   = row & (T_DIM - 1);
  const int id  = idx[row];
  const int c   = threadIdx.x * 4;
  float4 a = *(const float4*)(tok + (size_t)id * E_DIM + c);
  float4 p = *(const float4*)(pos + (size_t)t  * E_DIM + c);
  float4 o; o.x = a.x + p.x; o.y = a.y + p.y; o.z = a.z + p.z; o.w = a.w + p.w;
  *(float4*)(x + (size_t)row * E_DIM + c) = o;
}

// ---------------------------------------------------------------------------
// LayerNorm over E=1024, one row per block (256 threads, 4 floats/thread)
// ---------------------------------------------------------------------------
__global__ __launch_bounds__(256)
void ln_kernel(const float* __restrict__ x, const float* __restrict__ g,
               const float* __restrict__ bt, float* __restrict__ y) {
  __shared__ float red[256];
  const int row = blockIdx.x, tid = threadIdx.x;
  const float* xr = x + (size_t)row * E_DIM;
  float4 v = *(const float4*)(xr + tid * 4);

  red[tid] = v.x + v.y + v.z + v.w;
  __syncthreads();
  for (int off = 128; off > 0; off >>= 1) {
    if (tid < off) red[tid] += red[tid + off];
    __syncthreads();
  }
  const float mu = red[0] * (1.0f / E_DIM);
  __syncthreads();

  float dx = v.x - mu, dy = v.y - mu, dz = v.z - mu, dw = v.w - mu;
  red[tid] = dx * dx + dy * dy + dz * dz + dw * dw;
  __syncthreads();
  for (int off = 128; off > 0; off >>= 1) {
    if (tid < off) red[tid] += red[tid + off];
    __syncthreads();
  }
  const float rstd = rsqrtf(red[0] * (1.0f / E_DIM) + 1e-5f);

  float4 gg = *(const float4*)(g  + tid * 4);
  float4 bb = *(const float4*)(bt + tid * 4);
  float4 o;
  o.x = dx * rstd * gg.x + bb.x;
  o.y = dy * rstd * gg.y + bb.y;
  o.z = dz * rstd * gg.z + bb.z;
  o.w = dw * rstd * gg.w + bb.w;
  *(float4*)(y + (size_t)row * E_DIM + tid * 4) = o;
}

// ---------------------------------------------------------------------------
// WMMA GEMM: C[M,N] = A[M,K] @ W[N,K]^T, f32 in/out, bf16 math via LDS stage.
// Block tile 128x128x32, 256 threads = 8 waves (2x4), wave tile 64x32.
// Software pipelined: next k-tile global loads issued before the WMMA phase
// so VMEM latency overlaps matrix math.
// EPI: 0 none, 1 +res, 2 +bias,GELU, 3 +bias,+res, 4 none+nontemporal store.
// All problem dims divide the tiles (M=4096; N,K in {1024,3072,4096,32000}).
// ---------------------------------------------------------------------------
#define LDSTRIDE 40 // 32 bf16 + pad, 80B row (16B multiple, bank-spread)

template <int EPI>
__global__ __launch_bounds__(256)
void gemm_kernel(const float* __restrict__ A, const float* __restrict__ W,
                 const float* __restrict__ bias, const float* res,
                 float* __restrict__ C, int M, int N, int K) {
  __shared__ __align__(16) unsigned short lA[128 * LDSTRIDE];
  __shared__ __align__(16) unsigned short lB[128 * LDSTRIDE];

  const int tid  = threadIdx.x;
  const int lane = tid & 31, wave = tid >> 5;
  const int hf   = lane >> 4, l16 = lane & 15;
  const int wm   = wave >> 2, wn = wave & 3;      // 2x4 wave grid
  const int mbase = blockIdx.y * 128, nbase = blockIdx.x * 128;

  floatx8 acc[4][2];
  const floatx8 zero = {0.f, 0.f, 0.f, 0.f, 0.f, 0.f, 0.f, 0.f};
  for (int mt = 0; mt < 4; ++mt)
    for (int nt = 0; nt < 2; ++nt) acc[mt][nt] = zero;

  // Per-thread staging slots: 4 float4 per tile per matrix.
  float4 ra[4], rw[4];
#pragma unroll
  for (int i = 0; i < 4; ++i) {
    const int slot = tid + i * 256;               // 0..1023 float4 slots
    const int row  = slot >> 3;
    const int kq   = (slot & 7) * 4;
    ra[i] = *(const float4*)(A + (size_t)(mbase + row) * K + kq);
    rw[i] = *(const float4*)(W + (size_t)(nbase + row) * K + kq);
  }

  for (int k0 = 0; k0 < K; k0 += 32) {
    // Convert staged registers -> LDS bf16 tiles.
#pragma unroll
    for (int i = 0; i < 4; ++i) {
      const int slot = tid + i * 256;
      const int row  = slot >> 3;
      const int kq   = (slot & 7) * 4;
      uint2 pa; pa.x = pk2(ra[i].x, ra[i].y); pa.y = pk2(ra[i].z, ra[i].w);
      *(uint2*)(&lA[row * LDSTRIDE + kq]) = pa;
      uint2 pw; pw.x = pk2(rw[i].x, rw[i].y); pw.y = pk2(rw[i].z, rw[i].w);
      *(uint2*)(&lB[row * LDSTRIDE + kq]) = pw;
    }
    __syncthreads();

    // Issue next k-tile global loads now; their wait lands at the next
    // iteration's convert phase, hidden behind the WMMAs below.
    if (k0 + 32 < K) {
#pragma unroll
      for (int i = 0; i < 4; ++i) {
        const int slot = tid + i * 256;
        const int row  = slot >> 3;
        const int kq   = (slot & 7) * 4;
        ra[i] = *(const float4*)(A + (size_t)(mbase + row) * K + k0 + 32 + kq);
        rw[i] = *(const float4*)(W + (size_t)(nbase + row) * K + k0 + 32 + kq);
      }
    }
    if (k0 + 64 < K) { // warm L2 two tiles ahead (global_prefetch_b8)
      __builtin_prefetch(W + (size_t)(nbase + (tid >> 1)) * K + k0 + 64, 0, 1);
      __builtin_prefetch(A + (size_t)(mbase + (tid >> 1)) * K + k0 + 64, 0, 1);
    }

    bf16x16 af[4], bfr[2];
#pragma unroll
    for (int mt = 0; mt < 4; ++mt)
      af[mt] = frag_ld(lA, LDSTRIDE, wm * 64 + mt * 16 + l16, 0, hf);
#pragma unroll
    for (int nt = 0; nt < 2; ++nt)
      bfr[nt] = frag_ld(lB, LDSTRIDE, wn * 32 + nt * 16 + l16, 0, hf);
#pragma unroll
    for (int mt = 0; mt < 4; ++mt)
#pragma unroll
      for (int nt = 0; nt < 2; ++nt)
        acc[mt][nt] = wmma_bf16(af[mt], bfr[nt], acc[mt][nt]);
    __syncthreads();
  }

  // Epilogue + store. C-layout per ISA 7.12.2: vgpr i holds row
  // (lane<16 ? i : i+8), col = lane&15 within the 16x16 tile.
#pragma unroll
  for (int mt = 0; mt < 4; ++mt) {
#pragma unroll
    for (int nt = 0; nt < 2; ++nt) {
      const int col = nbase + wn * 32 + nt * 16 + l16;
#pragma unroll
      for (int i = 0; i < 8; ++i) {
        const int row = mbase + wm * 64 + mt * 16 + hf * 8 + i;
        float v = acc[mt][nt][i];
        if (EPI == 2 || EPI == 3) v += bias[col];
        if (EPI == 2) v = 0.5f * v * (1.0f + erff(v * 0.70710678118654752f));
        if (EPI == 1 || EPI == 3) v += res[(size_t)row * N + col];
        if (EPI == 4)
          __builtin_nontemporal_store(v, C + (size_t)row * N + col);
        else
          C[(size_t)row * N + col] = v;
      }
    }
  }
}

// ---------------------------------------------------------------------------
// Flash attention: grid = B*H*(T/64) blocks, 128 threads = 4 independent
// waves; each wave owns 16 queries, streams 32-key blocks with online
// softmax. qkv layout: [B*T, 3E] with q|k|v at column offsets 0|E|2E,
// head h at h*D within each. Output written to out[B*T, E] (f32).
// ---------------------------------------------------------------------------
#define QSTRIDE 72 // 64 bf16 + pad -> 144B rows (16B multiple)
#define PSTRIDE 40 // 32 bf16 + pad -> 80B rows

__global__ __launch_bounds__(128)
void attn_kernel(const float* __restrict__ qkv, float* __restrict__ out) {
  __shared__ __align__(16) unsigned short sQ [4][16 * QSTRIDE];
  __shared__ __align__(16) unsigned short sK [4][32 * QSTRIDE];
  __shared__ __align__(16) unsigned short sVt[4][64 * PSTRIDE];
  __shared__ __align__(16) unsigned short sP [4][16 * PSTRIDE];

  const int wave = threadIdx.x >> 5, lane = threadIdx.x & 31;
  const int hf = lane >> 4, l16 = lane & 15;
  const int qt = blockIdx.x & 15;       // T/64 = 16 query tiles
  const int bh = blockIdx.x >> 4;
  const int b  = bh >> 4;               // H = 16
  const int h  = bh & 15;
  const int qbase = qt * 64 + wave * 16;
  const float scale = 0.125f;           // 1/sqrt(D)

  unsigned short* lQ  = sQ[wave];
  unsigned short* lK  = sK[wave];
  unsigned short* lVt = sVt[wave];
  unsigned short* lP  = sP[wave];

  const size_t rs = 3 * E_DIM;
  const float* qp = qkv + (size_t)b * T_DIM * rs + h * D_DIM;
  const float* kp = qp + E_DIM;
  const float* vp = qp + 2 * E_DIM;

  // Stage Q tile (16x64) once.
#pragma unroll
  for (int i = 0; i < 8; ++i) {
    const int slot = lane + i * 32;     // 256 float4 slots
    const int r = slot >> 4, dq = (slot & 15) * 4;
    float4 a = *(const float4*)(qp + (size_t)(qbase + r) * rs + dq);
    uint2 p; p.x = pk2(a.x, a.y); p.y = pk2(a.z, a.w);
    *(uint2*)(&lQ[r * QSTRIDE + dq]) = p;
  }
  const bf16x16 qa0 = frag_ld(lQ, QSTRIDE, l16, 0,  hf);
  const bf16x16 qa1 = frag_ld(lQ, QSTRIDE, l16, 32, hf);

  const floatx8 zero = {0.f, 0.f, 0.f, 0.f, 0.f, 0.f, 0.f, 0.f};
  floatx8 o[4] = {zero, zero, zero, zero};
  float rmax[8], rsum[8];
#pragma unroll
  for (int i = 0; i < 8; ++i) { rmax[i] = -__builtin_inff(); rsum[i] = 0.0f; }

  const int send = qbase + 15;
  for (int s0 = 0; s0 <= send; s0 += 32) {
    // Stage K (32x64, row-major) and V transposed (64 d-rows x 32 keys).
#pragma unroll
    for (int i = 0; i < 16; ++i) {
      const int slot = lane + i * 32;   // 512 float4 slots
      const int r = slot >> 4, dq = (slot & 15) * 4;
      float4 a = *(const float4*)(kp + (size_t)(s0 + r) * rs + dq);
      uint2 p; p.x = pk2(a.x, a.y); p.y = pk2(a.z, a.w);
      *(uint2*)(&lK[r * QSTRIDE + dq]) = p;
      float4 v = *(const float4*)(vp + (size_t)(s0 + r) * rs + dq);
      lVt[(dq + 0) * PSTRIDE + r] = f2bf(v.x);
      lVt[(dq + 1) * PSTRIDE + r] = f2bf(v.y);
      lVt[(dq + 2) * PSTRIDE + r] = f2bf(v.z);
      lVt[(dq + 3) * PSTRIDE + r] = f2bf(v.w);
    }

    // S = Q @ K^T over D=64 (two k-steps), keys split into two 16-col tiles.
    floatx8 S0 = zero, S1 = zero;
    S0 = wmma_bf16(qa0, frag_ld(lK, QSTRIDE, l16,       0,  hf), S0);
    S0 = wmma_bf16(qa1, frag_ld(lK, QSTRIDE, l16,       32, hf), S0);
    S1 = wmma_bf16(qa0, frag_ld(lK, QSTRIDE, 16 + l16,  0,  hf), S1);
    S1 = wmma_bf16(qa1, frag_ld(lK, QSTRIDE, 16 + l16,  32, hf), S1);

    // Online softmax: rows of a 16x16 C tile live in lane halves; all
    // cross-lane reductions stay inside 16-lane groups -> DPP16 butterflies.
#pragma unroll
    for (int i = 0; i < 8; ++i) {
      const int row = qbase + hf * 8 + i;
      float s0v = S0[i] * scale;
      float s1v = S1[i] * scale;
      if (s0 + l16      > row) s0v = -__builtin_inff();
      if (s0 + 16 + l16 > row) s1v = -__builtin_inff();
      const float bm = redmax16(fmaxf(s0v, s1v));
      const float nm = fmaxf(rmax[i], bm);
      const float sc = __expf(rmax[i] - nm);   // exp(-inf)=0 on first block
      const float p0 = __expf(s0v - nm);
      const float p1 = __expf(s1v - nm);
      const float prs = redsum16(p0 + p1);
      rsum[i] = rsum[i] * sc + prs;
      rmax[i] = nm;
      o[0][i] *= sc; o[1][i] *= sc; o[2][i] *= sc; o[3][i] *= sc;
      lP[(hf * 8 + i) * PSTRIDE + l16]      = f2bf(p0);
      lP[(hf * 8 + i) * PSTRIDE + 16 + l16] = f2bf(p1);
    }

    // O += P(16x32) @ V(32x64): V^T rows are d-indices -> B fragments.
    const bf16x16 pa = frag_ld(lP, PSTRIDE, l16, 0, hf);
#pragma unroll
    for (int dt = 0; dt < 4; ++dt)
      o[dt] = wmma_bf16(pa, frag_ld(lVt, PSTRIDE, dt * 16 + l16, 0, hf), o[dt]);
  }

  // Normalize and write y back at (b*T+row, h*D+d).
#pragma unroll
  for (int i = 0; i < 8; ++i) {
    const int row = qbase + hf * 8 + i;
    const float inv = 1.0f / rsum[i];
#pragma unroll
    for (int dt = 0; dt < 4; ++dt)
      out[((size_t)b * T_DIM + row) * E_DIM + h * D_DIM + dt * 16 + l16] =
          o[dt][i] * inv;
  }
}

// ---------------------------------------------------------------------------
// Host-side orchestration (all on `stream`, graph-capture safe).
// Workspace layout (floats): X[4096*1024] | H[4096*1024] | QKV[4096*3072] |
// MID[4096*4096]  => 37,748,736 floats = 151 MB.
// ---------------------------------------------------------------------------
extern "C" void kernel_launch(void* const* d_in, const int* in_sizes, int n_in,
                              void* d_out, int out_size, void* d_ws, size_t ws_size,
                              hipStream_t stream) {
  (void)in_sizes; (void)n_in; (void)out_size; (void)ws_size;
  const int*   idx    = (const int*)  d_in[0];
  const float* tok    = (const float*)d_in[1];
  const float* pos    = (const float*)d_in[2];
  const float* qkv_w  = (const float*)d_in[3];
  const float* proj_w = (const float*)d_in[4];
  const float* ln1_g  = (const float*)d_in[5];
  const float* ln1_b  = (const float*)d_in[6];
  const float* ln2_g  = (const float*)d_in[7];
  const float* ln2_b  = (const float*)d_in[8];
  const float* fc1_w  = (const float*)d_in[9];
  const float* fc1_b  = (const float*)d_in[10];
  const float* fc2_w  = (const float*)d_in[11];
  const float* fc2_b  = (const float*)d_in[12];
  const float* lnf_g  = (const float*)d_in[13];
  const float* lnf_b  = (const float*)d_in[14];
  const float* head_w = (const float*)d_in[15];
  float* logits = (float*)d_out;

  float* X    = (float*)d_ws;
  float* Hbuf = X    + (size_t)M_ROWS * E_DIM;
  float* QKV  = Hbuf + (size_t)M_ROWS * E_DIM;
  float* MID  = QKV  + (size_t)M_ROWS * 3 * E_DIM;

  embed_kernel<<<M_ROWS, 256, 0, stream>>>(idx, tok, pos, X);

  for (int l = 0; l < L_DIM; ++l) {
    const float* qw  = qkv_w  + (size_t)l * 3 * E_DIM * E_DIM;
    const float* pw  = proj_w + (size_t)l * E_DIM * E_DIM;
    const float* f1w = fc1_w  + (size_t)l * 4 * E_DIM * E_DIM;
    const float* f1b = fc1_b  + (size_t)l * 4 * E_DIM;
    const float* f2w = fc2_w  + (size_t)l * E_DIM * 4 * E_DIM;
    const float* f2b = fc2_b  + (size_t)l * E_DIM;

    // h = LN1(x); qkv = h @ qkv_w^T
    ln_kernel<<<M_ROWS, 256, 0, stream>>>(X, ln1_g + l * E_DIM, ln1_b + l * E_DIM, Hbuf);
    gemm_kernel<0><<<dim3(3 * E_DIM / 128, M_ROWS / 128), 256, 0, stream>>>(
        Hbuf, qw, nullptr, nullptr, QKV, M_ROWS, 3 * E_DIM, E_DIM);

    // attn -> Hbuf (reuses LN1 buffer)
    attn_kernel<<<B_DIM * H_DIM * (T_DIM / 64), 128, 0, stream>>>(QKV, Hbuf);

    // x = x + attn @ proj_w^T (in-place residual)
    gemm_kernel<1><<<dim3(E_DIM / 128, M_ROWS / 128), 256, 0, stream>>>(
        Hbuf, pw, nullptr, X, X, M_ROWS, E_DIM, E_DIM);

    // h = LN2(x); mid = gelu(h @ fc1^T + b1); x = x + mid @ fc2^T + b2
    ln_kernel<<<M_ROWS, 256, 0, stream>>>(X, ln2_g + l * E_DIM, ln2_b + l * E_DIM, Hbuf);
    gemm_kernel<2><<<dim3(4 * E_DIM / 128, M_ROWS / 128), 256, 0, stream>>>(
        Hbuf, f1w, f1b, nullptr, MID, M_ROWS, 4 * E_DIM, E_DIM);
    gemm_kernel<3><<<dim3(E_DIM / 128, M_ROWS / 128), 256, 0, stream>>>(
        MID, f2w, f2b, X, X, M_ROWS, E_DIM, 4 * E_DIM);
  }

  // Final LN + LM head -> logits [4096, 32000]; non-temporal store epilogue
  // keeps the 524MB logits stream from evicting head_w (131MB) out of L2.
  ln_kernel<<<M_ROWS, 256, 0, stream>>>(X, lnf_g, lnf_b, Hbuf);
  gemm_kernel<4><<<dim3(32000 / 128, M_ROWS / 128), 256, 0, stream>>>(
      Hbuf, head_w, nullptr, nullptr, logits, M_ROWS, 32000, E_DIM);
}
